// AttnRNNDecoder_2662879723919
// MI455X (gfx1250) — matrix operand order repaired and checked by reading, not compiled, not run on hardware
//
#include <hip/hip_runtime.h>
#include <hip/hip_bf16.h>

// Problem dims (match reference)
#define B_   32
#define L_   512
#define DE_  256
#define H__  256
#define DO__ 256
#define T_   128
#define TS_  127      // T-1 timesteps
#define NC_  8000
#define NB_  128
#define V__  8128

typedef __attribute__((ext_vector_type(16))) __bf16 v16bf;
typedef __attribute__((ext_vector_type(8)))  float  v8f;

// ---------------------------------------------------------------------------
// bf16 helpers + WMMA fragment plumbing (contiguous-K fragment order, wave32)
// ---------------------------------------------------------------------------
__device__ __forceinline__ unsigned short f2bf(float f) {
  unsigned int u = __builtin_bit_cast(unsigned int, f);
  u += 0x7FFFu + ((u >> 16) & 1u);          // round-to-nearest-even
  return (unsigned short)(u >> 16);
}

union FragU { v16bf v; unsigned int u[8]; };

// A fragment 16x32 (MxK) from LDS row-major bf16:
// lane&15 = row M; lanes 0-15 hold K=0..15, lanes 16-31 hold K=16..31
__device__ __forceinline__ v16bf frag_a_from_lds(const unsigned short* p, int ldk) {
  int lane = threadIdx.x & 31;
  int row  = lane & 15;
  int kg   = lane >> 4;
  FragU f;
#pragma unroll
  for (int i = 0; i < 8; ++i) {
    int k0 = (i << 1) + (kg << 4);
    f.u[i] = *(const unsigned int*)(p + row * ldk + k0);
  }
  return f.v;
}

// B fragment 32x16 (KxN) from LDS row-major W (N x K): lane&15 = col N;
// lanes 0-15 hold K=0..15, lanes 16-31 hold K=16..31 (so C = A * W^T).
__device__ __forceinline__ v16bf frag_b_from_lds(const unsigned short* p, int ldk) {
  int lane = threadIdx.x & 31;
  int col  = lane & 15;
  int kh   = lane >> 4;
  FragU f;
#pragma unroll
  for (int i = 0; i < 8; ++i) {
    int k0 = (i << 1) + (kh << 4);
    f.u[i] = *(const unsigned int*)(p + col * ldk + k0);
  }
  return f.v;
}

__device__ __forceinline__ v8f wmma_bf16(v16bf a, v16bf b, v8f c) {
  return __builtin_amdgcn_wmma_f32_16x16x32_bf16(false, a, false, b, (short)0, c,
                                                 false, false);
}

// ---------------------------------------------------------------------------
// Pack f32 matrix into fragment-linear bf16 tiles (tile = 16 rows x 32 K;
// stored tile-major as 32 lanes x 8 dwords contiguous -> frag load = 2x b128).
// LAYOUT 0: A-fragment order       (row = M)
// LAYOUT 1: A-fragment, transposed source (element (row,k) = src[k*lda+row])
// LAYOUT 2: B-fragment order       (row = N; B k0 packing)
// Rows >= Rvalid pack as zero (tile padding keeps GEMM loops guard-free).
// Two sources (concat along K at 'split') support the fused [W_ih | W_hh].
// ---------------------------------------------------------------------------
template <int LAYOUT>
__global__ __launch_bounds__(256) void pack_kernel(
    unsigned int* __restrict__ dst,
    const float* __restrict__ srcA, int lda,
    const float* __restrict__ srcB, int ldb, int split,
    int R, int Rvalid, int C, long srcStrideZ, long dstStrideZ) {
  int z = blockIdx.z;
  const float* sA = srcA + (long)z * srcStrideZ;
  const float* sB = srcB + (long)z * srcStrideZ;
  unsigned int* d = dst + (long)z * dstStrideZ;
  int KT = C >> 5;
  int tile = blockIdx.x * 8 + (threadIdx.x >> 5);
  if (tile >= (R >> 4) * KT) return;
  int nt = tile / KT, kt = tile - nt * KT;
  int lane = threadIdx.x & 31;
  int row = (nt << 4) + (lane & 15);
  int kg = lane >> 4;
  bool ok = row < Rvalid;
  unsigned int* o = d + ((long)tile << 8) + (lane << 3);
#pragma unroll
  for (int i = 0; i < 8; ++i) {
    int k0;
    if (LAYOUT == 2) k0 = (kt << 5) + (i << 1) + (kg << 4);
    else             k0 = (kt << 5) + (i << 1) + (kg << 4);
    float f0 = 0.f, f1 = 0.f;
    if (LAYOUT == 1) {
      f0 = sA[(long)k0 * lda + row];
      f1 = sA[(long)(k0 + 1) * lda + row];
    } else if (ok) {
      f0 = (k0 < split) ? sA[(long)row * lda + k0]
                        : sB[(long)row * ldb + (k0 - split)];
      f1 = (k0 + 1 < split) ? sA[(long)row * lda + k0 + 1]
                            : sB[(long)row * ldb + (k0 + 1 - split)];
    }
    o[i] = (unsigned int)f2bf(f0) | ((unsigned int)f2bf(f1) << 16);
  }
}

// ---------------------------------------------------------------------------
// WMMA GEMV core: y(16*NT per wave interleaved) = W(N x K) * x(K).
// W pre-swizzled fragment tiles on the A side (full utilization);
// x (bf16 pairs in LDS) occupies column 0 of the B fragment (lanes 0/16).
// Wave 'w' owns output tiles nt = w + 8*j.  D column 0 -> lanes 0 & 16.
// ---------------------------------------------------------------------------
template <int NT, int KT>
__device__ __forceinline__ void gemv_wmma(const unsigned int* __restrict__ Wsw,
                                          const unsigned short* __restrict__ xbf,
                                          v8f* acc) {
  const int lane = threadIdx.x & 31;
  const int wave = threadIdx.x >> 5;
  const int col = lane & 15, kh = lane >> 4;
#pragma unroll 2
  for (int kt = 0; kt < KT; ++kt) {
    FragU bx;
    if (col == 0) {
#pragma unroll
      for (int v = 0; v < 8; ++v)
        bx.u[v] = *(const unsigned int*)(xbf + (kt << 5) + (v << 1) + (kh << 4));
    } else {
#pragma unroll
      for (int v = 0; v < 8; ++v) bx.u[v] = 0u;
    }
#pragma unroll
    for (int j = 0; j < NT; ++j) {
      int nt = wave + (j << 3);
      const unsigned int* p = Wsw + ((long)(nt * KT + kt) << 8) + (lane << 3);
      FragU a;
#pragma unroll
      for (int i = 0; i < 8; ++i) a.u[i] = p[i];
      acc[j] = wmma_bf16(a.v, bx.v, acc[j]);
    }
  }
}

// ---------------------------------------------------------------------------
// LDS-staged bf16-WMMA GEMM:  C(MxN) = A(MxK) * W(NxK)^T (+ epilogue)
// MODE 0: seq_q plain row-major store (+bias)
// MODE 2: per-batch logits vs batched encodings -> out[... + 8000 + n] + mask
// ---------------------------------------------------------------------------
#define GM 64
#define GN 128
#define GK 32
#define LDK 34

template <int MODE>
__global__ __launch_bounds__(256) void gemm_bf16_kernel(
    const float* __restrict__ A, long a_row_stride, int M,
    const float* __restrict__ W, int N, int K,
    const float* __restrict__ bias,
    const unsigned char* __restrict__ mask,
    float* __restrict__ out) {
  __shared__ __align__(16) unsigned short Asm[GM * LDK];
  __shared__ __align__(16) unsigned short Wsm[GN * LDK];

  const int tid = threadIdx.x;
  const int m0 = blockIdx.x * GM;
  const int n0 = blockIdx.y * GN;
  const int bz = blockIdx.z;

  const float* Ab = A;
  const float* Wb = W;
  if (MODE == 2) {
    Ab = A + bz * DO__;
    Wb = W + (long)bz * NB_ * DO__;
  }

  const int wid = tid >> 5;
  const int wm = wid & 3;
  const int wn = wid >> 2;

  v8f acc[4] = {};

  for (int kc = 0; kc < K; kc += GK) {
    for (int e = tid; e < GM * GK; e += 256) {
      int r = e >> 5, c = e & 31;
      int gm = m0 + r;
      float v = (gm < M) ? Ab[(long)gm * a_row_stride + kc + c] : 0.f;
      Asm[r * LDK + c] = f2bf(v);
    }
    for (int e = tid; e < GN * GK; e += 256) {
      int r = e >> 5, c = e & 31;
      int gn = n0 + r;
      float v = (gn < N) ? Wb[(long)gn * K + kc + c] : 0.f;
      Wsm[r * LDK + c] = f2bf(v);
    }
    __syncthreads();
    v16bf af = frag_a_from_lds(Asm + wm * 16 * LDK, LDK);
#pragma unroll
    for (int j = 0; j < 4; ++j) {
      v16bf bf = frag_b_from_lds(Wsm + (wn * 64 + j * 16) * LDK, LDK);
      acc[j] = wmma_bf16(af, bf, acc[j]);
    }
    __syncthreads();
  }

  const int lane = tid & 31;
  const int nloc = lane & 15;
  const int mb = (lane >> 4) * 8;
#pragma unroll
  for (int j = 0; j < 4; ++j) {
#pragma unroll
    for (int v = 0; v < 8; ++v) {
      int m = m0 + wm * 16 + mb + v;
      int n = n0 + wn * 64 + j * 16 + nloc;
      if (m < M && n < N) {
        float val = acc[j][v];
        if (MODE == 0) {
          out[(long)m * DE_ + n] = val + bias[n];
        } else {
          float madd = mask[bz * NB_ + n] ? 0.f : -3.0e38f;
          out[(long)(bz * TS_ + m) * V__ + NC_ + n] = val + madd;
        }
      }
    }
  }
}

// ---------------------------------------------------------------------------
// LDS-free packed GEMM for the dominant logits-vs-common-embedding product.
// A = packed lin tiles (254 x 8); B = packed common tiles PADDED to 512 x 8
// (rows >= 8000 are zero), so the compute loop is branch-free: the kt-loop is
// pure b128 frag loads + chained WMMAs.  Only stores check nt < 500.
// ---------------------------------------------------------------------------
__global__ __launch_bounds__(256) void logits_common_packed_kernel(
    const unsigned int* __restrict__ Asw,
    const unsigned int* __restrict__ Bsw,
    float* __restrict__ out) {
  const int mt = blockIdx.x;
  const int wave = threadIdx.x >> 5;
  const int lane = threadIdx.x & 31;
  const int ntb = blockIdx.y * 32 + wave * 4;   // padded tile space [0,512)
  const int NTILES = NC_ / 16;                  // 500 valid tiles

  v8f acc[4] = {};
#pragma unroll 1
  for (int kt = 0; kt < 8; ++kt) {
    FragU a;
    const unsigned int* pa = Asw + ((long)(mt * 8 + kt) << 8) + (lane << 3);
#pragma unroll
    for (int i = 0; i < 8; ++i) a.u[i] = pa[i];
#pragma unroll
    for (int j = 0; j < 4; ++j) {
      FragU bb;
      const unsigned int* pb = Bsw + ((long)((ntb + j) * 8 + kt) << 8) + (lane << 3);
#pragma unroll
      for (int i = 0; i < 8; ++i) bb.u[i] = pb[i];
      acc[j] = wmma_bf16(a.v, bb.v, acc[j]);
    }
  }

  const int nloc = lane & 15;
  const int mrow = (lane >> 4) << 3;
#pragma unroll
  for (int j = 0; j < 4; ++j) {
    int nt = ntb + j;
    if (nt >= NTILES) continue;                 // store-side guard only
#pragma unroll
    for (int v = 0; v < 8; ++v) {
      int m = (mt << 4) + mrow + v;             // m = t*32 + b
      int n = (nt << 4) + nloc;
      int t = m >> 5, b = m & 31;
      out[(long)(b * TS_ + t) * V__ + n] = acc[j][v];
    }
  }
}

// ---------------------------------------------------------------------------
// Gather teacher-forced embeddings: prev[(t*32+b)][d]
// ---------------------------------------------------------------------------
__global__ __launch_bounds__(256) void gather_kernel(
    const int* __restrict__ tgt, const float* __restrict__ common,
    const float* __restrict__ batched, float* __restrict__ prev) {
  int bid = blockIdx.x;            // t*32 + b
  int t = bid >> 5, b = bid & 31;
  int idx = tgt[b * T_ + t];
  const float* src;
  if (idx < NC_) {
    int ci = idx < 0 ? 0 : (idx > NC_ - 1 ? NC_ - 1 : idx);
    src = common + (long)ci * DO__;
  } else {
    int bi = idx - NC_;
    if (bi > NB_ - 1) bi = NB_ - 1;
    src = batched + ((long)b * NB_ + bi) * DO__;
  }
  prev[(long)bid * DO__ + threadIdx.x] = src[threadIdx.x];
}

// ---------------------------------------------------------------------------
// Recurrent core: one WG per batch element, 127 sequential steps.
// Every GEMV runs on WMMA via pre-swizzled bf16 weight tiles.
// ---------------------------------------------------------------------------
__device__ __forceinline__ float sigm(float x) { return 1.f / (1.f + __expf(-x)); }

__global__ __launch_bounds__(256) void recurrent_kernel(
    const unsigned char* __restrict__ emsk,      // (32,512)
    const unsigned int* __restrict__ seqqSW,     // per b: 32x8 tiles (512x256)
    const unsigned int* __restrict__ encTSW,     // per b: 16x16 tiles (256x512)
    const float* __restrict__ prev,              // (127,32,256) f32
    const unsigned int* __restrict__ WkSW,       // 16x16 tiles (256x512)
    const float* __restrict__ kb,
    const unsigned int* __restrict__ WcSW,       // 16x16 tiles (256x512)
    const float* __restrict__ cb,
    const unsigned int* __restrict__ WgSW,       // 64x16 tiles (1024x512)
    const float* __restrict__ bih, const float* __restrict__ bhh,
    const unsigned int* __restrict__ WoSW,       // 16x8 tiles (256x256)
    const float* __restrict__ ob,
    float* __restrict__ lin) {                   // (127,32,256) f32
  const int b = blockIdx.x;
  const int tid = threadIdx.x;
  const int lane = tid & 31;
  const int wave = tid >> 5;
  const int m8 = (lane >> 4) << 3;               // D col-0 row offset

  __shared__ __align__(4) unsigned short xk_bf[512];  // [pe ; h]
  __shared__ __align__(4) unsigned short xc_bf[512];  // [pe ; attn]
  __shared__ __align__(4) unsigned short xg_bf[512];  // [comb ; h]
  __shared__ __align__(4) unsigned short key_bf[256];
  __shared__ __align__(4) unsigned short w_bf[512];
  __shared__ float sc[512];
  __shared__ float g[1024];
  __shared__ float cst[256];
  __shared__ float red[256];

  cst[tid] = 0.f;
  xk_bf[256 + tid] = 0;        // h = 0
  xg_bf[256 + tid] = 0;
  const unsigned int* sqb = seqqSW + (long)b * (32 * 8 * 256);
  const unsigned int* etb = encTSW + (long)b * (16 * 16 * 256);
  __syncthreads();

  for (int t = 0; t < TS_; ++t) {
    unsigned short pb = f2bf(prev[((long)t * B_ + b) * DO__ + tid]);
    xk_bf[tid] = pb;
    xc_bf[tid] = pb;
    __syncthreads();

    // key = Wk @ [pe; h] + kb   (N=256, K=512)
    {
      v8f acc[2] = {};
      gemv_wmma<2, 16>(WkSW, xk_bf, acc);
      if ((lane & 15) == 0) {
#pragma unroll
        for (int j = 0; j < 2; ++j) {
          int n0 = (wave + (j << 3)) << 4;
#pragma unroll
          for (int v = 0; v < 8; ++v) {
            int n = n0 + m8 + v;
            key_bf[n] = f2bf(acc[j][v] + kb[n]);
          }
        }
      }
    }
    __syncthreads();

    // scores[l] = seq_q[b] @ key   (M=512 l-tiles on A side, K=256)
    {
      v8f acc[4] = {};
      gemv_wmma<4, 8>(sqb, key_bf, acc);
      if ((lane & 15) == 0) {
#pragma unroll
        for (int j = 0; j < 4; ++j) {
          int l0 = (wave + (j << 3)) << 4;
#pragma unroll
          for (int v = 0; v < 8; ++v) {
            int l = l0 + m8 + v;
            float val = acc[j][v];
            if (!emsk[b * L_ + l]) val = -3.0e38f;
            sc[l] = val;
          }
        }
      }
    }
    __syncthreads();

    // softmax over 512
    float lm = fmaxf(sc[tid], sc[tid + 256]);
    red[tid] = lm;
    __syncthreads();
    for (int s = 128; s > 0; s >>= 1) {
      if (tid < s) red[tid] = fmaxf(red[tid], red[tid + s]);
      __syncthreads();
    }
    float mx = red[0];
    __syncthreads();
    float e0 = __expf(sc[tid] - mx), e1 = __expf(sc[tid + 256] - mx);
    red[tid] = e0 + e1;
    __syncthreads();
    for (int s = 128; s > 0; s >>= 1) {
      if (tid < s) red[tid] += red[tid + s];
      __syncthreads();
    }
    float inv = 1.f / red[0];
    w_bf[tid] = f2bf(e0 * inv);
    w_bf[tid + 256] = f2bf(e1 * inv);
    __syncthreads();

    // attn = enc^T @ w   (A = encT d-tiles, K=512)
    {
      v8f acc[2] = {};
      gemv_wmma<2, 16>(etb, w_bf, acc);
      if ((lane & 15) == 0) {
#pragma unroll
        for (int j = 0; j < 2; ++j) {
          int n0 = (wave + (j << 3)) << 4;
#pragma unroll
          for (int v = 0; v < 8; ++v)
            xc_bf[256 + n0 + m8 + v] = f2bf(acc[j][v]);
        }
      }
    }
    __syncthreads();

    // comb = relu(Wc @ [pe; attn] + cb)
    {
      v8f acc[2] = {};
      gemv_wmma<2, 16>(WcSW, xc_bf, acc);
      if ((lane & 15) == 0) {
#pragma unroll
        for (int j = 0; j < 2; ++j) {
          int n0 = (wave + (j << 3)) << 4;
#pragma unroll
          for (int v = 0; v < 8; ++v) {
            int n = n0 + m8 + v;
            xg_bf[n] = f2bf(fmaxf(acc[j][v] + cb[n], 0.f));
          }
        }
      }
    }
    __syncthreads();

    // gates = [Wih|Whh] @ [comb; h] + biases   (N=1024, K=512)
    {
      v8f acc[8] = {};
      gemv_wmma<8, 16>(WgSW, xg_bf, acc);
      if ((lane & 15) == 0) {
#pragma unroll
        for (int j = 0; j < 8; ++j) {
          int n0 = (wave + (j << 3)) << 4;
#pragma unroll
          for (int v = 0; v < 8; ++v) {
            int n = n0 + m8 + v;
            g[n] = acc[j][v] + bih[n] + bhh[n];
          }
        }
      }
    }
    __syncthreads();

    // LSTM elementwise update
    {
      float cn = sigm(g[256 + tid]) * cst[tid] + sigm(g[tid]) * tanhf(g[512 + tid]);
      float hn = sigm(g[768 + tid]) * tanhf(cn);
      cst[tid] = cn;
      unsigned short hb = f2bf(hn);
      xk_bf[256 + tid] = hb;
      xg_bf[256 + tid] = hb;
    }
    __syncthreads();

    // lin = Wo @ h + ob -> workspace
    {
      v8f acc[2] = {};
      gemv_wmma<2, 8>(WoSW, xg_bf + 256, acc);
      if ((lane & 15) == 0) {
#pragma unroll
        for (int j = 0; j < 2; ++j) {
          int n0 = (wave + (j << 3)) << 4;
#pragma unroll
          for (int v = 0; v < 8; ++v) {
            int n = n0 + m8 + v;
            lin[((long)t * B_ + b) * DO__ + n] = acc[j][v] + ob[n];
          }
        }
      }
    }
    __syncthreads();
  }
}

// ---------------------------------------------------------------------------
// In-place log-softmax over each 8128-wide row of the output
// ---------------------------------------------------------------------------
__global__ __launch_bounds__(256) void logsoftmax_kernel(float* __restrict__ out) {
  float* p = out + (long)blockIdx.x * V__;
  __shared__ float red[256];
  int tid = threadIdx.x;

  float lm = -3.0e38f;
  for (int i = tid; i < V__; i += 256) lm = fmaxf(lm, p[i]);
  red[tid] = lm;
  __syncthreads();
  for (int s = 128; s > 0; s >>= 1) {
    if (tid < s) red[tid] = fmaxf(red[tid], red[tid + s]);
    __syncthreads();
  }
  float mx = red[0];
  __syncthreads();

  float ls = 0.f;
  for (int i = tid; i < V__; i += 256) ls += __expf(p[i] - mx);
  red[tid] = ls;
  __syncthreads();
  for (int s = 128; s > 0; s >>= 1) {
    if (tid < s) red[tid] += red[tid + s];
    __syncthreads();
  }
  float d = mx + logf(red[0]);
  __syncthreads();
  for (int i = tid; i < V__; i += 256) p[i] = p[i] - d;
}

// ---------------------------------------------------------------------------
// Host orchestration
// ---------------------------------------------------------------------------
// Workspace layout (4-byte element offsets):
#define WS_SEQQ   0L          // f32 seq_q (16384 x 256)        = 4194304
#define WS_PREV   4194304L    // f32 prev  (127,32,256)         = 1040384
#define WS_LIN    5234688L    // f32 lin   (127,32,256)         = 1040384
#define WU_WK     6275072L    // packed Wk   (256x512)  tiles   = 65536
#define WU_WC     6340608L    // packed Wc   (256x512)          = 65536
#define WU_WG     6406144L    // packed [Wih|Whh] (1024x512)    = 262144
#define WU_WO     6668288L    // packed Wo   (256x256)          = 32768
#define WU_SEQQ   6701056L    // packed seq_q (16384x256)       = 2097152
#define WU_ENCT   8798208L    // packed enc^T (32 x 256x512)    = 2097152
#define WU_LINSW  10895360L   // packed lin A-tiles (4064x256)  = 520192
#define WU_CMNSW  11415552L   // packed common B-tiles (8192pad x256) = 1048576
// total 12464128 * 4B ~= 49.9 MB

extern "C" void kernel_launch(void* const* d_in, const int* in_sizes, int n_in,
                              void* d_out, int out_size, void* d_ws, size_t ws_size,
                              hipStream_t stream) {
  const float* enc          = (const float*)d_in[0];
  const unsigned char* emsk = (const unsigned char*)d_in[1];
  const float* batched      = (const float*)d_in[2];
  const unsigned char* bmsk = (const unsigned char*)d_in[3];
  const int*   tgt          = (const int*)d_in[4];
  const float* common       = (const float*)d_in[5];
  const float* Wk           = (const float*)d_in[6];
  const float* kb           = (const float*)d_in[7];
  const float* Wq           = (const float*)d_in[8];
  const float* qb           = (const float*)d_in[9];
  const float* Wc           = (const float*)d_in[10];
  const float* cb           = (const float*)d_in[11];
  const float* Wih          = (const float*)d_in[12];
  const float* Whh          = (const float*)d_in[13];
  const float* bih          = (const float*)d_in[14];
  const float* bhh          = (const float*)d_in[15];
  const float* Wo           = (const float*)d_in[16];
  const float* ob           = (const float*)d_in[17];

  float* out = (float*)d_out;
  float* ws  = (float*)d_ws;
  unsigned int* wsu = (unsigned int*)d_ws;

  float* seqq = ws + WS_SEQQ;
  float* prev = ws + WS_PREV;
  float* lin  = ws + WS_LIN;
  unsigned int* WkSW   = wsu + WU_WK;
  unsigned int* WcSW   = wsu + WU_WC;
  unsigned int* WgSW   = wsu + WU_WG;
  unsigned int* WoSW   = wsu + WU_WO;
  unsigned int* seqqSW = wsu + WU_SEQQ;
  unsigned int* encTSW = wsu + WU_ENCT;
  unsigned int* linSW  = wsu + WU_LINSW;
  unsigned int* cmnSW  = wsu + WU_CMNSW;

  // 1) Gather teacher-forced embeddings -> prev (127,32,256)
  gather_kernel<<<dim3(TS_ * B_), 256, 0, stream>>>(tgt, common, batched, prev);

  // 2) seq_q = enc @ Wq^T + qb (WMMA bf16), plain row-major f32
  gemm_bf16_kernel<0><<<dim3((B_ * L_) / GM, DE_ / GN, 1), 256, 0, stream>>>(
      enc, DE_, B_ * L_, Wq, DE_, DE_, qb, nullptr, seqq);

  // 3) Pack weights / seq_q / enc^T / common-embedding into fragment tiles
  pack_kernel<0><<<dim3(32, 1, 1), 256, 0, stream>>>(
      WkSW, Wk, 512, Wk, 512, 512, 256, 256, 512, 0, 0);
  pack_kernel<0><<<dim3(32, 1, 1), 256, 0, stream>>>(
      WcSW, Wc, 512, Wc, 512, 512, 256, 256, 512, 0, 0);
  pack_kernel<0><<<dim3(128, 1, 1), 256, 0, stream>>>(
      WgSW, Wih, 256, Whh, 256, 256, 1024, 1024, 512, 0, 0);
  pack_kernel<0><<<dim3(16, 1, 1), 256, 0, stream>>>(
      WoSW, Wo, 256, Wo, 256, 256, 256, 256, 256, 0, 0);
  pack_kernel<0><<<dim3(1024, 1, 1), 256, 0, stream>>>(
      seqqSW, seqq, 256, seqq, 256, 256, B_ * L_, B_ * L_, 256, 0, 0);
  pack_kernel<1><<<dim3(32, 1, B_), 256, 0, stream>>>(
      encTSW, enc, 256, enc, 256, 512, 256, 256, 512,
      (long)L_ * DE_, (long)16 * 16 * 256);
  pack_kernel<2><<<dim3(512, 1, 1), 256, 0, stream>>>(     // padded to 8192 rows
      cmnSW, common, 256, common, 256, 256, 8192, NC_, 256, 0, 0);

  // 4) Sequential recurrence: 32 independent WGs, all GEMVs on WMMA
  recurrent_kernel<<<dim3(B_), 256, 0, stream>>>(
      emsk, seqqSW, encTSW, prev, WkSW, kb, WcSW, cb, WgSW, bih, bhh, WoSW, ob,
      lin);

  // 5) Pack lin into A-fragment tiles (254 x 8 tiles)
  pack_kernel<0><<<dim3(254, 1, 1), 256, 0, stream>>>(
      linSW, lin, 256, lin, 256, 256, TS_ * B_, TS_ * B_, 256, 0, 0);

  // 6) Dominant logits GEMM (4064 x 8000 x 256): LDS-free, branch-free WMMA
  logits_common_packed_kernel<<<dim3(254, 16, 1), 256, 0, stream>>>(
      linSW, cmnSW, out);

  // 7) logits vs per-batch encodings: 32 x (127 x 128 x 256) WMMA GEMMs + mask
  gemm_bf16_kernel<2><<<dim3((TS_ + GM - 1) / GM, (NB_ + GN - 1) / GN, B_),
                        256, 0, stream>>>(
      lin, (long)B_ * DO__, TS_, batched, NB_, DO__, nullptr, bmsk, out);

  // 8) In-place log-softmax over each 8128-wide row
  logsoftmax_kernel<<<dim3(TS_ * B_), 256, 0, stream>>>(out);
}
